// RITS2_14001593384978
// MI455X (gfx1250) — matrix operand order repaired and checked by reading, not compiled, not run on hardware
//
#include <hip/hip_runtime.h>

// -------------------------------------------------------------------------
// RITS/GRU-D fused LSTM for MI455X (gfx1250, wave32, WMMA f16 16x16x32).
// Batch rows are independent through the recurrence -> each workgroup owns
// 64 rows for all 20 timesteps. h and c live in LDS; weights (f16 concat,
// 3.9 MB) are L2-resident. Compute-bound -> f16 WMMA with f32 accumulate.
// -------------------------------------------------------------------------

typedef __attribute__((ext_vector_type(16))) _Float16 v16h;
typedef __attribute__((ext_vector_type(8)))  float    v8f;

#define SEQ     20
#define NF      812
#define HID     400
#define VAR     811
#define BATCH   8192
#define KIH     813            // NF + 1 (mask column)
#define KTOT    1216           // 813 + 400 padded to 38*32
#define NGATE   1600           // 4*HID
#define MROWS   64             // batch rows per workgroup
#define NWAVES  10
#define NTHREADS 320
#define JT      25             // HID/16 hidden-unit tiles
#define PPW     10             // (4 mtiles * 25 jtiles) / 10 waves

__device__ __forceinline__ float sigmoidf_(float x) {
    return 1.0f / (1.0f + __expf(-x));
}
__device__ __forceinline__ float tanhf_(float x) {
    float e = __expf(2.0f * x);
    return 1.0f - 2.0f / (e + 1.0f);
}

// ---------------- precompute: f16 weight concat [W_ih | W_hh | 0pad] -----
__global__ void prep_wcat(const float* __restrict__ W_ih,
                          const float* __restrict__ W_hh,
                          _Float16* __restrict__ Wcat) {
    int idx = blockIdx.x * blockDim.x + threadIdx.x;
    if (idx >= NGATE * KTOT) return;
    int n = idx / KTOT;
    int k = idx - n * KTOT;
    float v;
    if (k < KIH)            v = W_ih[n * KIH + k];
    else if (k < KIH + HID) v = W_hh[n * HID + (k - KIH)];
    else                    v = 0.0f;
    Wcat[idx] = (_Float16)v;
}

// ---------------- precompute: w_sum, bias sum, zero loss accumulators ----
__global__ void prep_misc(const float* __restrict__ W_decay,
                          const float* __restrict__ b_ih,
                          const float* __restrict__ b_hh,
                          float* __restrict__ wsum,
                          float* __restrict__ bias,
                          float* __restrict__ lnum,
                          float* __restrict__ lden) {
    int idx = blockIdx.x * blockDim.x + threadIdx.x;
    if (idx < NGATE) bias[idx] = b_ih[idx] + b_hh[idx];
    if (idx < HID) {
        float s = 0.0f;
        for (int c = 0; c < NF; ++c) s += W_decay[idx * NF + c];
        wsum[idx] = s;
    }
    if (idx < SEQ) { lnum[idx] = 0.0f; lden[idx] = 0.0f; }
}

// ---------------- finalize: loss scalar ----------------------------------
__global__ void finalize_loss(const float* __restrict__ lnum,
                              const float* __restrict__ lden,
                              float* __restrict__ out) {
    if (blockIdx.x == 0 && threadIdx.x == 0) {
        float s = 0.0f;
        for (int t = 0; t < SEQ; ++t) s += lnum[t] / (lden[t] + 1e-5f);
        out[0] = s / (float)SEQ;
    }
}

// ---------------- main fused recurrence ----------------------------------
__global__ void __launch_bounds__(NTHREADS, 1)
rits_fused(const float* __restrict__ values,
           const float* __restrict__ masks,
           const float* __restrict__ deltas,
           const float* __restrict__ b_decay,
           const float* __restrict__ W_reg,
           const float* __restrict__ b_reg,
           const _Float16* __restrict__ Wcat,
           const float* __restrict__ wsum,
           const float* __restrict__ bias_sum,
           float* __restrict__ lnum_g,
           float* __restrict__ lden_g,
           float* __restrict__ out) {
    extern __shared__ char smem[];
    _Float16* Ash = (_Float16*)smem;                                  // [64][1216] f16
    _Float16* Hsc = (_Float16*)(smem + (size_t)MROWS * KTOT * 2);     // [64][400]  f16
    float*    Csc = (float*)(smem + (size_t)MROWS * KTOT * 2
                                  + (size_t)MROWS * HID * 2);         // [64][400]  f32
    float* d_sh   = Csc + MROWS * HID;
    float* m_sh   = d_sh + MROWS;
    float* xh_sh  = m_sh + MROWS;
    float* xvo_sh = xh_sh + MROWS;
    float* red_sh = xvo_sh + MROWS;                                   // [2]

    const int tid    = threadIdx.x;
    const int wave   = tid >> 5;
    const int lane   = tid & 31;
    const int laneHi = lane >> 4;       // 0 | 1
    const int lanelo = lane & 15;
    const int row0   = blockIdx.x * MROWS;

    // init: zero K-pad columns, h scratch, c scratch (h0 = c0 = 0)
    for (int i = tid; i < MROWS; i += NTHREADS) {
        Ash[i * KTOT + 1213] = (_Float16)0.0f;
        Ash[i * KTOT + 1214] = (_Float16)0.0f;
        Ash[i * KTOT + 1215] = (_Float16)0.0f;
    }
    for (int i = tid; i < MROWS * HID; i += NTHREADS) {
        Hsc[i] = (_Float16)0.0f;
        Csc[i] = 0.0f;
    }

    const float breg = b_reg[0];

    for (int t = 0; t < SEQ; ++t) {
        __syncthreads();  // previous GEMM reads / Hsc,Csc writes complete

        // stage per-row scalars for this step
        if (tid < MROWS) {
            int gr = row0 + tid;
            d_sh[tid] = deltas[(size_t)gr * SEQ + t];
            m_sh[tid] = masks[(size_t)gr * SEQ + t];
        }
        if (tid == MROWS) { red_sh[0] = 0.0f; red_sh[1] = 0.0f; }

        // stream x -> A (f16); hold out the VAR column
        for (int i = tid; i < MROWS * NF; i += NTHREADS) {
            int r   = i / NF;
            int col = i - r * NF;
            float v = values[((size_t)(row0 + r) * SEQ + t) * NF + col];
            if (col == VAR) xvo_sh[r] = v;
            else            Ash[r * KTOT + col] = (_Float16)v;
        }
        __syncthreads();  // d_sh ready

        // decay h and publish to A's h-region (K columns 813..1212)
        for (int i = tid; i < MROWS * HID; i += NTHREADS) {
            int r = i / HID;
            int j = i - r * HID;
            float h = (float)Hsc[i];
            float a = d_sh[r] * wsum[j] + b_decay[j];
            float g = __expf(-fmaxf(a, 0.0f));
            Ash[r * KTOT + KIH + j] = (_Float16)(h * g);
        }
        __syncthreads();  // decayed h visible

        // x_h[r] = h_decayed . W_reg + b_reg   (per-wave row reduction)
        for (int r = wave; r < MROWS; r += NWAVES) {
            float acc = 0.0f;
            for (int j = lane; j < HID; j += 32)
                acc += (float)Ash[r * KTOT + KIH + j] * W_reg[j];
            for (int off = 16; off > 0; off >>= 1)
                acc += __shfl_xor(acc, off, 32);
            if (lane == 0) xh_sh[r] = acc + breg;
        }
        __syncthreads();

        // imputation, VAR/mask columns, loss partials
        if (tid < MROWS) {
            float m  = m_sh[tid];
            float xh = xh_sh[tid];
            float xv = xvo_sh[tid] * m + (1.0f - m) * xh;
            Ash[tid * KTOT + VAR] = (_Float16)xv;
            Ash[tid * KTOT + NF]  = (_Float16)m;     // mask column (K=812)
            out[1 + (size_t)(row0 + tid) * SEQ + t] = xv;
            atomicAdd(&red_sh[0], fabsf(xv - xh) * m);
            atomicAdd(&red_sh[1], m);
        }
        __syncthreads();  // A fully built; loss partials done
        if (tid == 0) {
            atomicAdd(&lnum_g[t], red_sh[0]);
            atomicAdd(&lden_g[t], red_sh[1]);
        }

        // prefetch next timestep's x while the GEMM runs
        if (t + 1 < SEQ) {
            const int LINES = 26;  // ceil(812*4 / 128)
            for (int i = tid; i < MROWS * LINES; i += NTHREADS) {
                int r  = i / LINES;
                int ln = i - r * LINES;
                const char* p = (const char*)(values +
                    ((size_t)(row0 + r) * SEQ + (t + 1)) * NF) + ln * 128;
                __builtin_prefetch(p, 0, 1);
            }
        }

        // ---- fused GEMM + LSTM: each wave owns 10 (mtile,jtile) pairs ----
        // and computes all four gate tiles per pair, so the i/f/g/o coupling
        // and the cell update happen in-register (no gates LDS round-trip).
#pragma unroll 1
        for (int p = 0; p < PPW; ++p) {
            const int pair  = wave * PPW + p;
            const int mtile = pair / JT;          // 0..3
            const int jtile = pair - mtile * JT;  // 0..24
            const int rloc  = mtile * 16 + lanelo;  // A row this lane holds
            const int jn    = jtile * 16 + lanelo;  // hidden-unit column

            // A layout: lanes 0-15 K=0..15 contig, lanes 16-31 K=16..31
            const _Float16* arow = Ash + (size_t)rloc * KTOT + (laneHi ? 16 : 0);
            // B layout (same k order as A): lanes 0-15 K=0..15 contig, lanes 16-31 K=16..31
            const _Float16* b0 = Wcat + (size_t)(0 * HID + jn) * KTOT + (laneHi ? 16 : 0);
            const _Float16* b1 = Wcat + (size_t)(1 * HID + jn) * KTOT + (laneHi ? 16 : 0);
            const _Float16* b2 = Wcat + (size_t)(2 * HID + jn) * KTOT + (laneHi ? 16 : 0);
            const _Float16* b3 = Wcat + (size_t)(3 * HID + jn) * KTOT + (laneHi ? 16 : 0);

            v8f acc_i, acc_f, acc_g, acc_o;
#pragma unroll
            for (int e = 0; e < 8; ++e) {
                acc_i[e] = 0.0f; acc_f[e] = 0.0f; acc_g[e] = 0.0f; acc_o[e] = 0.0f;
            }

            for (int k = 0; k < KTOT / 32; ++k) {
                v16h a;
                *(uint4*)&a       = *(const uint4*)(arow + k * 32);
                *((uint4*)&a + 1) = *(const uint4*)(arow + k * 32 + 8);
                v16h w0, w1, w2, w3;
                *(uint4*)&w0       = *(const uint4*)(b0 + k * 32);
                *((uint4*)&w0 + 1) = *(const uint4*)(b0 + k * 32 + 8);
                *(uint4*)&w1       = *(const uint4*)(b1 + k * 32);
                *((uint4*)&w1 + 1) = *(const uint4*)(b1 + k * 32 + 8);
                *(uint4*)&w2       = *(const uint4*)(b2 + k * 32);
                *((uint4*)&w2 + 1) = *(const uint4*)(b2 + k * 32 + 8);
                *(uint4*)&w3       = *(const uint4*)(b3 + k * 32);
                *((uint4*)&w3 + 1) = *(const uint4*)(b3 + k * 32 + 8);

                acc_i = __builtin_amdgcn_wmma_f32_16x16x32_f16(false, a, false, w0,
                        (short)0, acc_i, false, false);
                acc_f = __builtin_amdgcn_wmma_f32_16x16x32_f16(false, a, false, w1,
                        (short)0, acc_f, false, false);
                acc_g = __builtin_amdgcn_wmma_f32_16x16x32_f16(false, a, false, w2,
                        (short)0, acc_g, false, false);
                acc_o = __builtin_amdgcn_wmma_f32_16x16x32_f16(false, a, false, w3,
                        (short)0, acc_o, false, false);
            }

            const float bi = bias_sum[0 * HID + jn];
            const float bf = bias_sum[1 * HID + jn];
            const float bg = bias_sum[2 * HID + jn];
            const float bo = bias_sum[3 * HID + jn];

#pragma unroll
            for (int e = 0; e < 8; ++e) {
                // D layout: vgpr e -> M = e + 8*laneHi, lane&15 -> N
                int mrow = mtile * 16 + e + laneHi * 8;
                int cidx = mrow * HID + jn;
                float ig = sigmoidf_(acc_i[e] + bi);
                float fg = sigmoidf_(acc_f[e] + bf);
                float gg = tanhf_(acc_g[e] + bg);
                float og = sigmoidf_(acc_o[e] + bo);
                float cn = fg * Csc[cidx] + ig * gg;
                Csc[cidx] = cn;
                Hsc[cidx] = (_Float16)(og * tanhf_(cn));
            }
        }
    }
}

// -------------------------------------------------------------------------
extern "C" void kernel_launch(void* const* d_in, const int* in_sizes, int n_in,
                              void* d_out, int out_size, void* d_ws, size_t ws_size,
                              hipStream_t stream) {
    (void)in_sizes; (void)n_in; (void)out_size; (void)ws_size;
    const float* values  = (const float*)d_in[0];
    const float* masks   = (const float*)d_in[1];
    const float* deltas  = (const float*)d_in[2];
    const float* W_decay = (const float*)d_in[3];
    const float* b_decay = (const float*)d_in[4];
    const float* W_reg   = (const float*)d_in[5];
    const float* b_reg   = (const float*)d_in[6];
    const float* W_ih    = (const float*)d_in[7];
    const float* W_hh    = (const float*)d_in[8];
    const float* b_ih    = (const float*)d_in[9];
    const float* b_hh    = (const float*)d_in[10];
    float* out = (float*)d_out;

    // scratch layout
    char* ws = (char*)d_ws;
    _Float16* Wcat = (_Float16*)ws;                                  // 1600*1216 f16
    float* wsum = (float*)(ws + (size_t)NGATE * KTOT * 2);
    float* bias = wsum + HID;
    float* lnum = bias + NGATE;
    float* lden = lnum + SEQ;

    {
        int total = NGATE * KTOT;
        prep_wcat<<<(total + 255) / 256, 256, 0, stream>>>(W_ih, W_hh, Wcat);
    }
    prep_misc<<<(NGATE + 255) / 256, 256, 0, stream>>>(W_decay, b_ih, b_hh,
                                                       wsum, bias, lnum, lden);

    size_t lds = (size_t)MROWS * KTOT * 2      // A (f16)
               + (size_t)MROWS * HID * 2       // h scratch (f16)
               + (size_t)MROWS * HID * 4       // c state (f32)
               + (size_t)(4 * MROWS + 2) * 4;  // d/m/xh/xvar + reduction
    (void)hipFuncSetAttribute((const void*)rits_fused,
                              hipFuncAttributeMaxDynamicSharedMemorySize, (int)lds);
    rits_fused<<<BATCH / MROWS, NTHREADS, lds, stream>>>(
        values, masks, deltas, b_decay, W_reg, b_reg,
        Wcat, wsum, bias, lnum, lden, out);

    finalize_loss<<<1, 1, 0, stream>>>(lnum, lden, out);
}
